// SceneGraphConstructor_85907935855318
// MI455X (gfx1250) — matrix operand order repaired and checked by reading, not compiled, not run on hardware
//
#include <hip/hip_runtime.h>
#include <hip/hip_bf16.h>

typedef __attribute__((ext_vector_type(16))) __bf16  v16bf;
typedef __attribute__((ext_vector_type(8)))  float   v8f;
typedef __attribute__((ext_vector_type(4)))  float   v4f;
typedef __attribute__((ext_vector_type(2)))  float   v2f;
typedef __attribute__((ext_vector_type(4)))  unsigned v4u;

#define HW_      16384
#define NN_      128
#define CC_      768
#define NODE_IN_ 1545
#define NI_LD_   1568      // padded stride (49*32)
#define HID1_    512
#define DIM_     256
#define EDIM_    128

// workspace layout (float offsets)
#define INV_OFF 0
#define CX_OFF  1024
#define CY_OFF  2048
#define NI_OFF  4096
#define H_OFF   (NI_OFF + 1024 * NI_LD_)
#define HI_OFF  (H_OFF  + 1024 * HID1_)
#define HJ_OFF  (HI_OFF + 1024 * DIM_)
#define SQ_OFF  (HJ_OFF + 1024 * DIM_)
#define W1P_OFF (SQ_OFF + 1024)

// output layout (float offsets into d_out)
#define EDGE_OFF 262144L
#define ADJ_OFF  (262144L + 16777216L)

// ---------------------------------------------------------------------------
// helpers
// ---------------------------------------------------------------------------
__device__ __forceinline__ unsigned pk_bf16(float a, float b) {
  union { __bf16 h[2]; unsigned u; } x;
  x.h[0] = (__bf16)a; x.h[1] = (__bf16)b;
  return x.u;   // lowers to v_cvt_pk_bf16_f32
}

__device__ __forceinline__ unsigned lds_off(const void* p) {
  // generic LDS pointer: low 32 bits are the wave-relative LDS byte address
  return (unsigned)(unsigned long long)p;
}

__device__ __forceinline__ void async_load_b128(unsigned ldsaddr, const float* g) {
  asm volatile("global_load_async_to_lds_b128 %0, %1, off"
               :: "v"(ldsaddr), "v"((unsigned long long)g)
               : "memory");
}

__device__ __forceinline__ void wait_asynccnt0() {
  asm volatile("s_wait_asynccnt 0x0" ::: "memory");
}

// ---------------------------------------------------------------------------
// Kernel 1: per-(b,n) mask reductions: 1/(sum+eps), centroid cx, cy
// ---------------------------------------------------------------------------
__global__ __launch_bounds__(256) void mask_stats_kernel(
    const float* __restrict__ masks, float* __restrict__ inv,
    float* __restrict__ cx, float* __restrict__ cy) {
  int bn = blockIdx.x;
  const float* m = masks + (long)bn * HW_;
  int t = threadIdx.x;
  float s = 0.f, sx = 0.f, sy = 0.f;
  for (int i = t * 4; i < HW_; i += 1024) {
    v4f v = *(const v4f*)&m[i];
    int y = i >> 7, x = i & 127;
#pragma unroll
    for (int u = 0; u < 4; ++u) {
      s += v[u]; sx += v[u] * (float)(x + u); sy += v[u] * (float)y;
    }
  }
  __shared__ float rs[256], rx[256], ry[256];
  rs[t] = s; rx[t] = sx; ry[t] = sy;
  __syncthreads();
  for (int off = 128; off > 0; off >>= 1) {
    if (t < off) { rs[t] += rs[t + off]; rx[t] += rx[t + off]; ry[t] += ry[t + off]; }
    __syncthreads();
  }
  if (t == 0) {
    float tot = rs[0] + 1e-6f;
    inv[bn] = 1.f / tot;
    cx[bn] = rx[0] / tot;
    cy[bn] = ry[0] / tot;
  }
}

// ---------------------------------------------------------------------------
// Kernel 2: tail of node_input (+ zero the pad columns)
// ---------------------------------------------------------------------------
__global__ __launch_bounds__(256) void node_misc_kernel(
    const float* __restrict__ obj, const float* __restrict__ bboxes,
    const float* __restrict__ conf, const float* __restrict__ depth,
    const float* __restrict__ pose, const float* __restrict__ cx,
    const float* __restrict__ cy, float* __restrict__ node_input) {
  int bn = blockIdx.x;
  int b = bn >> 7;
  float* ni = node_input + (long)bn * NI_LD_;
  const float* of = obj + (long)bn * 769;
  for (int i = threadIdx.x; i < 769; i += 256) ni[768 + i] = of[i];
  for (int i = NODE_IN_ + threadIdx.x; i < NI_LD_; i += 256) ni[i] = 0.f;
  if (threadIdx.x == 0) {
    const float* bb = bboxes + bn * 4;
    float x1 = bb[0], y1 = bb[1], x2 = bb[2], y2 = bb[3];
    ni[1537] = 0.5f * (x1 + x2);
    ni[1538] = 0.5f * (y1 + y2);
    ni[1539] = x2 - x1;
    ni[1540] = y2 - y1;
    ni[1541] = conf[bn];
    float cxv = cx[bn], cyv = cy[bn];
    int cxi = (int)rintf(cxv); cxi = cxi < 0 ? 0 : (cxi > 127 ? 127 : cxi);
    int cyi = (int)rintf(cyv); cyi = cyi < 0 ? 0 : (cyi > 127 ? 127 : cyi);
    float z = depth[(long)b * HW_ + cyi * 128 + cxi];
    float X = cxv * z, Y = cyv * z;   // K = I3
    const float* P = pose + b * 16;
    float p0 = P[0]  * X + P[1]  * Y + P[2]  * z + P[3];
    float p1 = P[4]  * X + P[5]  * Y + P[6]  * z + P[7];
    float p2 = P[8]  * X + P[9]  * Y + P[10] * z + P[11];
    float p3 = P[12] * X + P[13] * Y + P[14] * z + P[15];
    ni[1542] = p0 / p3;
    ni[1543] = p1 / p3;
    ni[1544] = p2 / p3;
  }
}

// ---------------------------------------------------------------------------
// Kernel 3: pad node_w1 (512 x 1545) -> (512 x 1568), zero tail
// ---------------------------------------------------------------------------
__global__ __launch_bounds__(256) void pad_w1_kernel(const float* __restrict__ w1,
                                                     float* __restrict__ w1p) {
  long idx = (long)blockIdx.x * 256 + threadIdx.x;
  if (idx >= 512L * NI_LD_) return;
  int row = (int)(idx / NI_LD_);
  int k   = (int)(idx % NI_LD_);
  w1p[idx] = (k < NODE_IN_) ? w1[(long)row * NODE_IN_ + k] : 0.f;
}

// ---------------------------------------------------------------------------
// WMMA inner step shared by both GEMM variants
// ---------------------------------------------------------------------------
__device__ __forceinline__ void wmma_step(const __bf16 (*sA)[40], const __bf16 (*sB)[40],
                                          int mt, int nt0, int nt1, int r, int hf,
                                          v8f& acc0, v8f& acc1) {
  v16bf af;
  int arow = mt * 16 + r;
  int kb = hf * 16;
#pragma unroll
  for (int i = 0; i < 8; ++i) af[i] = sA[arow][kb + i];
#pragma unroll
  for (int i = 0; i < 8; ++i) af[8 + i] = sA[arow][kb + 8 + i];

  v16bf bf0, bf1;
  int kb2 = hf * 16;
  int bc0 = nt0 * 16 + r;
  int bc1 = nt1 * 16 + r;
#pragma unroll
  for (int i = 0; i < 16; ++i) bf0[i] = sB[bc0][kb2 + i];
#pragma unroll
  for (int i = 0; i < 16; ++i) bf1[i] = sB[bc1][kb2 + i];

  acc0 = __builtin_amdgcn_wmma_f32_16x16x32_bf16(false, af, false, bf0,
                                                 (short)0, acc0, false, false);
  acc1 = __builtin_amdgcn_wmma_f32_16x16x32_bf16(false, af, false, bf1,
                                                 (short)0, acc1, false, false);
}

// ---------------------------------------------------------------------------
// Generic NT-GEMM: out[row,col] = sum_k A[row,k]*Bt[col,k]
// K must be a multiple of 32; all rows 16B-aligned.
// ASYNC=1: f32 chunks staged with global_load_async_to_lds_b128 (double
//          buffered, next chunk overlapped with WMMA), then LDS-converted
//          to bf16.  ASYNC=0: synchronous b128 loads + packed bf16 stores.
// ---------------------------------------------------------------------------
template <int ASYNC>
__global__ __launch_bounds__(256) void gemm_nt_wmma_kernel(
    const float* __restrict__ A, int lda,
    const float* __restrict__ Bt, int ldb, long b_batch_stride, int rows_per_batch,
    int K,
    const float* __restrict__ bias,
    const float* __restrict__ rowscale,
    float* __restrict__ out, int ldo, int relu) {
  __shared__ __bf16 sA[64][40];
  __shared__ __bf16 sB[64][40];

  int t    = threadIdx.x;
  int wv   = t >> 5;
  int lane = t & 31;
  int r    = lane & 15;
  int hf   = lane >> 4;

  int m0 = blockIdx.x * 64;
  int n0 = blockIdx.y * 64;
  int batch = m0 / rows_per_batch;
  const float* Bp = Bt + (long)batch * b_batch_stride;

  int mt  = wv & 3;
  int nt0 = wv >> 2;
  int nt1 = nt0 + 2;

  // staging coords: 4 threads per row, 8 contiguous k each
  int srow = t >> 2;
  int kseg = (t & 3) * 8;

  v8f acc0 = {};
  v8f acc1 = {};

  if constexpr (ASYNC) {
    __shared__ float stA[2][64][32];
    __shared__ float stB[2][64][32];

    const float* aptr = &A [(long)(m0 + srow) * lda + kseg];
    const float* bptr = &Bp[(long)(n0 + srow) * ldb + kseg];

    // prologue: issue chunk 0
    async_load_b128(lds_off(&stA[0][srow][kseg]),     aptr);
    async_load_b128(lds_off(&stA[0][srow][kseg + 4]), aptr + 4);
    async_load_b128(lds_off(&stB[0][srow][kseg]),     bptr);
    async_load_b128(lds_off(&stB[0][srow][kseg + 4]), bptr + 4);

    int buf = 0;
    for (int k0 = 0; k0 < K; k0 += 32, buf ^= 1) {
      wait_asynccnt0();
      __syncthreads();
      // LDS f32 -> LDS bf16 (packed b128 store)
      {
        const float* a = &stA[buf][srow][kseg];
        const float* b = &stB[buf][srow][kseg];
        v4u pa = { pk_bf16(a[0], a[1]), pk_bf16(a[2], a[3]),
                   pk_bf16(a[4], a[5]), pk_bf16(a[6], a[7]) };
        v4u pb = { pk_bf16(b[0], b[1]), pk_bf16(b[2], b[3]),
                   pk_bf16(b[4], b[5]), pk_bf16(b[6], b[7]) };
        *(v4u*)&sA[srow][kseg] = pa;
        *(v4u*)&sB[srow][kseg] = pb;
      }
      __syncthreads();
      // overlap: issue next chunk before the WMMAs
      if (k0 + 32 < K) {
        const float* an = aptr + k0 + 32;
        const float* bn = bptr + k0 + 32;
        async_load_b128(lds_off(&stA[buf ^ 1][srow][kseg]),     an);
        async_load_b128(lds_off(&stA[buf ^ 1][srow][kseg + 4]), an + 4);
        async_load_b128(lds_off(&stB[buf ^ 1][srow][kseg]),     bn);
        async_load_b128(lds_off(&stB[buf ^ 1][srow][kseg + 4]), bn + 4);
      }
      wmma_step(sA, sB, mt, nt0, nt1, r, hf, acc0, acc1);
      __syncthreads();
    }
  } else {
    for (int k0 = 0; k0 < K; k0 += 32) {
      const float* a = &A [(long)(m0 + srow) * lda + k0 + kseg];
      const float* b = &Bp[(long)(n0 + srow) * ldb + k0 + kseg];
      v4f a0 = *(const v4f*)a;
      v4f a1 = *(const v4f*)(a + 4);
      v4f b0 = *(const v4f*)b;
      v4f b1 = *(const v4f*)(b + 4);
      if (k0 + 32 < K) __builtin_prefetch(a + 32, 0, 0);
      v4u pa = { pk_bf16(a0[0], a0[1]), pk_bf16(a0[2], a0[3]),
                 pk_bf16(a1[0], a1[1]), pk_bf16(a1[2], a1[3]) };
      v4u pb = { pk_bf16(b0[0], b0[1]), pk_bf16(b0[2], b0[3]),
                 pk_bf16(b1[0], b1[1]), pk_bf16(b1[2], b1[3]) };
      *(v4u*)&sA[srow][kseg] = pa;
      *(v4u*)&sB[srow][kseg] = pb;
      __syncthreads();
      wmma_step(sA, sB, mt, nt0, nt1, r, hf, acc0, acc1);
      __syncthreads();
    }
  }

  // epilogue: C layout — VGPR rr: row = rr + 8*(lane>=16); col = lane&15
#pragma unroll
  for (int s = 0; s < 2; ++s) {
    v8f acc = s ? acc1 : acc0;
    int nt = s ? nt1 : nt0;
#pragma unroll
    for (int rr = 0; rr < 8; ++rr) {
      int row = m0 + mt * 16 + rr + 8 * hf;
      int col = n0 + nt * 16 + r;
      float v = acc[rr];
      if (rowscale) v *= rowscale[row];
      if (bias) v += bias[col];
      if (relu) v = fmaxf(v, 0.f);
      out[(long)row * ldo + col] = v;
    }
  }
}

// ---------------------------------------------------------------------------
// Edge MLP: he = relu(hi[i] + hj[j] + b1) ; edge = he @ w2^T + b2
// block = (8 i's) x (16 j's) = 128 pairs x 128 cols, K=256 in chunks of 32.
// ---------------------------------------------------------------------------
__global__ __launch_bounds__(256) void edge_kernel(
    const float* __restrict__ hi, const float* __restrict__ hj,
    const float* __restrict__ eb1, const float* __restrict__ ew2,
    const float* __restrict__ eb2, float* __restrict__ out) {
  __shared__ float  s_hi[8][256];
  __shared__ float  s_hj[16][256];
  __shared__ __bf16 s_he[128][40];
  __shared__ __bf16 s_w2[128][40];

  int t    = threadIdx.x;
  int wv   = t >> 5;
  int lane = t & 31;
  int r    = lane & 15;
  int hf   = lane >> 4;

  int i0 = blockIdx.x * 8;
  int j0 = blockIdx.y * 16;
  int b  = blockIdx.z;

  for (int idx = t; idx < 8 * 256; idx += 256) {
    int n = idx >> 8, k = idx & 255;
    s_hi[n][k] = hi[((long)(b * NN_ + i0 + n)) * DIM_ + k] + eb1[k];
  }
  for (int idx = t; idx < 16 * 256; idx += 256) {
    int n = idx >> 8, k = idx & 255;
    s_hj[n][k] = hj[((long)(b * NN_ + j0 + n)) * DIM_ + k];
  }
  __syncthreads();

  v8f acc[8];
#pragma unroll
  for (int s = 0; s < 8; ++s) acc[s] = (v8f){};

  // staging coords: 2 threads per pair-row, 16 contiguous k each
  int srow = t >> 1;          // 0..127
  int ksg  = (t & 1) * 16;    // 0 or 16

  for (int kc = 0; kc < 8; ++kc) {
    int kbase = kc * 32;
    {
      int il = srow >> 4, jl = srow & 15;
      const float* hr = &s_hi[il][kbase + ksg];
      const float* jr = &s_hj[jl][kbase + ksg];
      const float* wr = &ew2[(long)srow * DIM_ + kbase + ksg];
      v4u ph, pw;
#pragma unroll
      for (int q = 0; q < 2; ++q) {
#pragma unroll
      for (int u = 0; u < 4; ++u) {
        float v0 = fmaxf(hr[8 * q + 2 * u] + jr[8 * q + 2 * u], 0.f);
        float v1 = fmaxf(hr[8 * q + 2 * u + 1] + jr[8 * q + 2 * u + 1], 0.f);
        ph[u] = pk_bf16(v0, v1);
        pw[u] = pk_bf16(wr[8 * q + 2 * u], wr[8 * q + 2 * u + 1]);
      }
      *(v4u*)&s_he[srow][ksg + 8 * q] = ph;
      *(v4u*)&s_w2[srow][ksg + 8 * q] = pw;
      }
    }
    __syncthreads();

    v16bf af;
    int prow = wv * 16 + r;
    int kb = hf * 16;
#pragma unroll
    for (int i = 0; i < 8; ++i) af[i] = s_he[prow][kb + i];
#pragma unroll
    for (int i = 0; i < 8; ++i) af[8 + i] = s_he[prow][kb + 8 + i];

#pragma unroll
    for (int s = 0; s < 8; ++s) {
      v16bf bfv;
      int col = s * 16 + r;
      int kb2 = hf * 16;
#pragma unroll
      for (int i = 0; i < 16; ++i) bfv[i] = s_w2[col][kb2 + i];
      acc[s] = __builtin_amdgcn_wmma_f32_16x16x32_bf16(false, af, false, bfv,
                                                       (short)0, acc[s], false, false);
    }
    __syncthreads();
  }

#pragma unroll
  for (int s = 0; s < 8; ++s) {
#pragma unroll
    for (int rr = 0; rr < 8; ++rr) {
      int pair = wv * 16 + rr + 8 * hf;
      int col  = s * 16 + r;
      int ig = i0 + (pair >> 4);
      int jg = j0 + (pair & 15);
      float v = acc[s][rr] + eb2[col];
      out[(((long)(b * NN_ + ig) * NN_) + jg) * EDIM_ + col] = v;
    }
  }
}

// ---------------------------------------------------------------------------
__global__ __launch_bounds__(256) void sq_kernel(const float* __restrict__ emb,
                                                 float* __restrict__ sq) {
  int idx = blockIdx.x * 256 + threadIdx.x;
  const float* e = emb + (long)idx * DIM_;
  float s = 0.f;
  for (int k = 0; k < DIM_; k += 4) {
    v4f v = *(const v4f*)&e[k];
    s += v[0] * v[0] + v[1] * v[1] + v[2] * v[2] + v[3] * v[3];
  }
  sq[idx] = s;
}

// ---------------------------------------------------------------------------
// adjacency: gram via v_wmma_f32_16x16x4_f32, then exp(-dist) epilogue
// ---------------------------------------------------------------------------
__global__ __launch_bounds__(32) void adj_kernel(const float* __restrict__ emb,
                                                 const float* __restrict__ sq,
                                                 float* __restrict__ out) {
  int lane = threadIdx.x & 31;
  int r = lane & 15, hf = lane >> 4;
  int i0 = blockIdx.x * 16;
  int j0 = blockIdx.y * 16;
  int b  = blockIdx.z;
  const float* eb = emb + (long)b * NN_ * DIM_;

  v8f acc = {};
  for (int k = 0; k < DIM_; k += 4) {
    v2f a, bb;
    int ks = k + hf * 2;
    a.x  = eb[(long)(i0 + r) * DIM_ + ks];
    a.y  = eb[(long)(i0 + r) * DIM_ + ks + 1];
    bb.x = eb[(long)(j0 + r) * DIM_ + ks];
    bb.y = eb[(long)(j0 + r) * DIM_ + ks + 1];
    acc = __builtin_amdgcn_wmma_f32_16x16x4_f32(false, a, false, bb,
                                                (short)0, acc, false, false);
  }
#pragma unroll
  for (int rr = 0; rr < 8; ++rr) {
    int ig = i0 + rr + 8 * hf;
    int jg = j0 + r;
    float d2 = sq[b * NN_ + ig] + sq[b * NN_ + jg] - 2.f * acc[rr];
    d2 = fmaxf(d2, 0.f);
    float d = (d2 > 0.f) ? sqrtf(d2) : 0.f;
    out[((long)(b * NN_ + ig)) * NN_ + jg] = __expf(-d);
  }
}

// ---------------------------------------------------------------------------
extern "C" void kernel_launch(void* const* d_in, const int* in_sizes, int n_in,
                              void* d_out, int out_size, void* d_ws, size_t ws_size,
                              hipStream_t stream) {
  const float* obj      = (const float*)d_in[0];
  const float* masks    = (const float*)d_in[1];
  const float* features = (const float*)d_in[2];
  const float* bboxes   = (const float*)d_in[3];
  const float* conf     = (const float*)d_in[4];
  const float* depth    = (const float*)d_in[5];
  const float* pose     = (const float*)d_in[6];
  const float* nw1      = (const float*)d_in[7];
  const float* nb1      = (const float*)d_in[8];
  const float* nw2      = (const float*)d_in[9];
  const float* nb2      = (const float*)d_in[10];
  const float* ew1      = (const float*)d_in[11];
  const float* eb1      = (const float*)d_in[12];
  const float* ew2      = (const float*)d_in[13];
  const float* eb2      = (const float*)d_in[14];

  float* out = (float*)d_out;
  float* ws  = (float*)d_ws;

  float* inv = ws + INV_OFF;
  float* cx  = ws + CX_OFF;
  float* cy  = ws + CY_OFF;
  float* ni  = ws + NI_OFF;
  float* h   = ws + H_OFF;
  float* hiw = ws + HI_OFF;
  float* hjw = ws + HJ_OFF;
  float* sqw = ws + SQ_OFF;
  float* w1p = ws + W1P_OFF;

  // 1) mask reductions
  mask_stats_kernel<<<1024, 256, 0, stream>>>(masks, inv, cx, cy);
  // 2) node_input tail + pad zeroing
  node_misc_kernel<<<1024, 256, 0, stream>>>(obj, bboxes, conf, depth, pose, cx, cy, ni);
  // 3) pad node_w1 into ws
  pad_w1_kernel<<<(512 * NI_LD_ + 255) / 256, 256, 0, stream>>>(nw1, w1p);
  // 4) pooled = m_norm @ feats^T -> node_input[:, 0:768]  (async staged)
  gemm_nt_wmma_kernel<1><<<dim3(16, 12), 256, 0, stream>>>(
      masks, HW_, features, HW_, (long)CC_ * HW_, NN_, HW_,
      nullptr, inv, ni, NI_LD_, 0);
  // 5) h = relu(node_input @ w1p^T + b1)  (async staged)
  gemm_nt_wmma_kernel<1><<<dim3(16, 8), 256, 0, stream>>>(
      ni, NI_LD_, w1p, NI_LD_, 0L, 1 << 30, NI_LD_,
      nb1, nullptr, h, HID1_, 1);
  // 6) node_emb = h @ w2^T + b2 -> d_out
  gemm_nt_wmma_kernel<0><<<dim3(16, 4), 256, 0, stream>>>(
      h, HID1_, nw2, HID1_, 0L, 1 << 30, HID1_,
      nb2, nullptr, out, DIM_, 0);
  // 7) hi / hj projections
  gemm_nt_wmma_kernel<0><<<dim3(16, 4), 256, 0, stream>>>(
      out, DIM_, ew1, 2 * DIM_, 0L, 1 << 30, DIM_,
      nullptr, nullptr, hiw, DIM_, 0);
  gemm_nt_wmma_kernel<0><<<dim3(16, 4), 256, 0, stream>>>(
      out, DIM_, ew1 + DIM_, 2 * DIM_, 0L, 1 << 30, DIM_,
      nullptr, nullptr, hjw, DIM_, 0);
  // 8) sum of squares
  sq_kernel<<<4, 256, 0, stream>>>(out, sqw);
  // 9) edge_emb
  edge_kernel<<<dim3(16, 8, 8), 256, 0, stream>>>(hiw, hjw, eb1, ew2, eb2, out + EDGE_OFF);
  // 10) adjacency
  adj_kernel<<<dim3(8, 8, 8), 32, 0, stream>>>(out, sqw, out + ADJ_OFF);
}
